// SelectiveSSM_35321811042500
// MI455X (gfx1250) — matrix operand order repaired and checked by reading, not compiled, not run on hardware
//
#include <hip/hip_runtime.h>
#include <hip/hip_bf16.h>

// ---------------------------------------------------------------------------
// SelectiveSSM for MI455X (gfx1250): bf16 WMMA GEMMs + f32 recurrence scan.
// D=1024, N=16, R=64, BSZ=4, S=4096  -> 16384 tokens.
// GEMMs: 64x64 tile per wave (4x4 WMMA fragments), 128x128 per 4-wave block.
//        __launch_bounds__(128,1) to unlock the VGPR budget (no spills).
// Scan: B/C staged to LDS with global_load_async_to_lds_b128 (ASYNCcnt).
// ---------------------------------------------------------------------------

#define Dm     1024
#define Nst    16
#define Rr     64
#define BSZ    4
#define SEQ    4096
#define MTOT   (BSZ * SEQ)        // 16384 tokens
#define XPW    128                // R + 4N

typedef __attribute__((ext_vector_type(16))) __bf16 bf16x16;
typedef __attribute__((ext_vector_type(8)))  float  v8f;

union FragU { bf16x16 v; uint4 q[2]; };

// ---- fragment loads per CDNA5 WMMA VGPR layouts (CDNA5 ISA §7.12.2) -------
// A-matrix 16x32 bf16: lane L holds row M=L%16; K-chunks [k0,k0+8) in v0..3
// and [k0+16,k0+24) in v4..7 where k0=(L/16)*8.
__device__ __forceinline__ bf16x16 load_frag_a(const __bf16* base, int lda, int lane) {
    const __bf16* p = base + (size_t)(lane & 15) * lda + ((lane >> 4) << 3);
    FragU u;
    u.q[0] = *(const uint4*)(p);
    u.q[1] = *(const uint4*)(p + 16);
    return u.v;
}
// B-matrix 32x16 bf16: lane L holds column N=L%16; K-chunks [k0,k0+8) and
// [k0+16,k0+24) where k0=(L/16)*8. Weight stored row-major (Nout x K).
__device__ __forceinline__ bf16x16 load_frag_b(const __bf16* base, int ldb, int lane) {
    const __bf16* p = base + (size_t)(lane & 15) * ldb + ((lane >> 4) << 3);
    FragU u;
    u.q[0] = *(const uint4*)(p);
    u.q[1] = *(const uint4*)(p + 16);
    return u.v;
}

__device__ __forceinline__ v8f wmma_bf16(bf16x16 a, bf16x16 b, v8f c) {
    // (neg_a, A, neg_b, B, c_mod, C, reuse_a, reuse_b)
    return __builtin_amdgcn_wmma_f32_16x16x32_bf16(false, a, false, b, (short)0, c,
                                                   false, false);
}

// One wave computes a 64x64 output tile (4x4 fragments): 16 WMMA per 16 b128.
// B-fragments are consumed one at a time to keep live fragments ~5, not 8.
__device__ __forceinline__ void gemm_tile_64x64(const __bf16* A, int lda,
                                                const __bf16* W, int ldb,
                                                int m0, int n0, int K, int lane,
                                                v8f c[4][4]) {
    for (int k = 0; k < K; k += 32) {
        bf16x16 a[4];
        #pragma unroll
        for (int i = 0; i < 4; ++i)
            a[i] = load_frag_a(A + (size_t)(m0 + 16 * i) * lda + k, lda, lane);
        #pragma unroll
        for (int j = 0; j < 4; ++j) {
            bf16x16 b = load_frag_b(W + (size_t)(n0 + 16 * j) * ldb + k, ldb, lane);
            #pragma unroll
            for (int i = 0; i < 4; ++i)
                c[i][j] = wmma_bf16(a[i], b, c[i][j]);
        }
    }
}

// Fast SiLU: x * rcp(1 + exp(-x))  (v_exp_f32 + v_rcp_f32, no IEEE div expansion)
__device__ __forceinline__ float silu_f(float x) {
    return x * __builtin_amdgcn_rcpf(1.0f + __expf(-x));
}
// Fast softplus with overflow guard (v_exp_f32 + v_log_f32)
__device__ __forceinline__ float softplus_f(float v) {
    return (v > 20.0f) ? v : __logf(1.0f + __expf(v));
}

// ---------------------------------------------------------------------------
// f32 -> bf16 conversion
// ---------------------------------------------------------------------------
__global__ void k_f32_to_bf16(const float* __restrict__ s, __bf16* __restrict__ d, int n) {
    int i = blockIdx.x * blockDim.x + threadIdx.x;
    if (i < n) d[i] = (__bf16)s[i];
}

// ---------------------------------------------------------------------------
// GEMM1: xz = x @ W_in.T  (M=16384, K=1024, N=2048), fused SiLU split.
// ---------------------------------------------------------------------------
__global__ __launch_bounds__(128, 1)
void k_gemm1(const __bf16* __restrict__ A, const __bf16* __restrict__ W,
             __bf16* __restrict__ xin, __bf16* __restrict__ zsil) {
    int lane = threadIdx.x & 31;
    int wave = threadIdx.x >> 5;
    int n0 = blockIdx.x * 128 + (wave & 1) * 64;
    int m0 = blockIdx.y * 128 + (wave >> 1) * 64;
    v8f c[4][4] = {};
    gemm_tile_64x64(A, Dm, W, Dm, m0, n0, Dm, lane, c);
    int mr = (lane >> 4) * 8, nc = lane & 15;
    #pragma unroll
    for (int i = 0; i < 4; ++i)
        #pragma unroll
        for (int j = 0; j < 4; ++j)
            #pragma unroll
            for (int r = 0; r < 8; ++r) {
                int m = m0 + 16 * i + mr + r;
                int n = n0 + 16 * j + nc;
                float v = silu_f(c[i][j][r]);
                if (n < Dm) xin [(size_t)m * Dm + n]        = (__bf16)v;
                else        zsil[(size_t)m * Dm + (n - Dm)] = (__bf16)v;
            }
}

// ---------------------------------------------------------------------------
// GEMM2: xp = xin @ W_x.T  (M=16384, K=1024, N=128) -> f32 (scan B/C) + bf16
// ---------------------------------------------------------------------------
__global__ __launch_bounds__(128, 1)
void k_gemm2(const __bf16* __restrict__ A, const __bf16* __restrict__ W,
             float* __restrict__ xp, __bf16* __restrict__ xp_bf) {
    int lane = threadIdx.x & 31;
    int wave = threadIdx.x >> 5;
    int n0 = (wave & 1) * 64;                       // N = 128 -> single block in x
    int m0 = blockIdx.y * 128 + (wave >> 1) * 64;
    v8f c[4][4] = {};
    gemm_tile_64x64(A, Dm, W, Dm, m0, n0, Dm, lane, c);
    int mr = (lane >> 4) * 8, nc = lane & 15;
    #pragma unroll
    for (int i = 0; i < 4; ++i)
        #pragma unroll
        for (int j = 0; j < 4; ++j)
            #pragma unroll
            for (int r = 0; r < 8; ++r) {
                int m = m0 + 16 * i + mr + r;
                int n = n0 + 16 * j + nc;
                float v = c[i][j][r];
                xp   [(size_t)m * XPW + n] = v;
                xp_bf[(size_t)m * XPW + n] = (__bf16)v;
            }
}

// ---------------------------------------------------------------------------
// GEMM3: dt = softplus(xp[:, :64] @ W_dt.T + b_dt)  (M=16384, K=64, N=1024)
// ---------------------------------------------------------------------------
__global__ __launch_bounds__(128, 1)
void k_gemm3(const __bf16* __restrict__ A, const __bf16* __restrict__ W,
             const float* __restrict__ b_dt, float* __restrict__ dt) {
    int lane = threadIdx.x & 31;
    int wave = threadIdx.x >> 5;
    int n0 = blockIdx.x * 128 + (wave & 1) * 64;
    int m0 = blockIdx.y * 128 + (wave >> 1) * 64;
    v8f c[4][4] = {};
    gemm_tile_64x64(A, XPW, W, Rr, m0, n0, Rr, lane, c);   // first 64 cols of xp
    int mr = (lane >> 4) * 8, nc = lane & 15;
    #pragma unroll
    for (int i = 0; i < 4; ++i)
        #pragma unroll
        for (int j = 0; j < 4; ++j)
            #pragma unroll
            for (int r = 0; r < 8; ++r) {
                int m = m0 + 16 * i + mr + r;
                int n = n0 + 16 * j + nc;
                dt[(size_t)m * Dm + n] = softplus_f(c[i][j][r] + b_dt[n]);
            }
}

// ---------------------------------------------------------------------------
// Scan: one thread per (b, d) channel; 16 complex states in registers.
// B/C (xp cols 64..127) staged in LDS via global_load_async_to_lds_b128
// (ASYNCcnt path), 32-step chunks (8 KB). Emits y * silu(z) as bf16.
// ---------------------------------------------------------------------------
__global__ __launch_bounds__(128)
void k_scan(const float* __restrict__ dt, const __bf16* __restrict__ xin,
            const float* __restrict__ xp, const __bf16* __restrict__ zsil,
            const float* __restrict__ A_log, __bf16* __restrict__ ygated) {
    const int b = blockIdx.y;
    const int d = blockIdx.x * 128 + threadIdx.x;

    float a_neg[Nst];
    #pragma unroll
    for (int n = 0; n < Nst; ++n) a_neg[n] = -__expf(A_log[n]);

    float h_re[Nst], h_im[Nst];
    #pragma unroll
    for (int n = 0; n < Nst; ++n) { h_re[n] = 0.0f; h_im[n] = 0.0f; }

    __shared__ float bc_s[32 * 64];   // 32 steps x (B[16x2] | C[16x2])
    const size_t tok0 = (size_t)b * SEQ;

    for (int c0 = 0; c0 < SEQ; c0 += 32) {
        __syncthreads();              // previous chunk fully consumed
        #pragma unroll
        for (int i = 0; i < 4; ++i) { // 512 x 16B by 128 threads, async to LDS
            int idx = threadIdx.x + i * 128;
            int row = idx >> 4;                   // 16 x b128 per step
            int col = (idx & 15) << 2;
            const float* gsrc = xp + (tok0 + c0 + row) * XPW + Rr + col;
            unsigned lds_dst = (unsigned)(size_t)(&bc_s[row * 64 + col]);
            asm volatile("global_load_async_to_lds_b128 %0, %1, off"
                         :: "v"(lds_dst), "v"(gsrc) : "memory");
        }
        asm volatile("s_wait_asynccnt 0" ::: "memory");
        __syncthreads();

        for (int tt = 0; tt < 32; ++tt) {
            const size_t tok = tok0 + c0 + tt;
            const float dtv = dt[tok * Dm + d];
            const float xv  = (float)xin[tok * Dm + d];
            const float gx  = dtv * xv;
            const float* Brow = &bc_s[tt * 64];
            float y = 0.0f;
            #pragma unroll
            for (int n = 0; n < Nst; ++n) {
                float alpha = __expf(dtv * a_neg[n]);
                float bre = Brow[2 * n], bim = Brow[2 * n + 1];
                float cre = Brow[32 + 2 * n], cim = Brow[32 + 2 * n + 1];
                h_re[n] = alpha * h_re[n] + gx * bre;
                h_im[n] = alpha * h_im[n] + gx * bim;
                y += cre * h_re[n] - cim * h_im[n];
            }
            float zs = (float)zsil[tok * Dm + d];
            ygated[tok * Dm + d] = (__bf16)(y * zs);
        }
    }
}

// ---------------------------------------------------------------------------
// GEMM4: out = ygated @ W_out.T + b_out  (M=16384, K=1024, N=1024) -> f32
// ---------------------------------------------------------------------------
__global__ __launch_bounds__(128, 1)
void k_gemm4(const __bf16* __restrict__ A, const __bf16* __restrict__ W,
             const float* __restrict__ b_out, float* __restrict__ out) {
    int lane = threadIdx.x & 31;
    int wave = threadIdx.x >> 5;
    int n0 = blockIdx.x * 128 + (wave & 1) * 64;
    int m0 = blockIdx.y * 128 + (wave >> 1) * 64;
    v8f c[4][4] = {};
    gemm_tile_64x64(A, Dm, W, Dm, m0, n0, Dm, lane, c);
    int mr = (lane >> 4) * 8, nc = lane & 15;
    #pragma unroll
    for (int i = 0; i < 4; ++i)
        #pragma unroll
        for (int j = 0; j < 4; ++j)
            #pragma unroll
            for (int r = 0; r < 8; ++r) {
                int m = m0 + 16 * i + mr + r;
                int n = n0 + 16 * j + nc;
                out[(size_t)m * Dm + n] = c[i][j][r] + b_out[n];
            }
}

// ---------------------------------------------------------------------------
// Host-side launcher
// ---------------------------------------------------------------------------
extern "C" void kernel_launch(void* const* d_in, const int* in_sizes, int n_in,
                              void* d_out, int out_size, void* d_ws, size_t ws_size,
                              hipStream_t stream) {
    (void)in_sizes; (void)n_in; (void)out_size; (void)ws_size;

    const float* x     = (const float*)d_in[0];   // (4,4096,1024)
    const float* W_in  = (const float*)d_in[1];   // (2048,1024)
    const float* W_x   = (const float*)d_in[2];   // (128,1024)
    const float* W_dt  = (const float*)d_in[3];   // (1024,64)
    const float* b_dt  = (const float*)d_in[4];   // (1024,)
    const float* W_out = (const float*)d_in[5];   // (1024,1024)
    const float* b_out = (const float*)d_in[6];   // (1024,)
    const float* A_log = (const float*)d_in[7];   // (16,)
    float* out = (float*)d_out;

    // Workspace layout (all offsets 256B-aligned)
    char* ws = (char*)d_ws;
    size_t off = 0;
    auto alloc = [&](size_t bytes) { char* p = ws + off; off += (bytes + 255) & ~(size_t)255; return p; };

    __bf16* x_bf    = (__bf16*)alloc((size_t)MTOT * Dm * 2);
    __bf16* Win_bf  = (__bf16*)alloc((size_t)2 * Dm * Dm * 2);
    __bf16* Wx_bf   = (__bf16*)alloc((size_t)XPW * Dm * 2);
    __bf16* Wdt_bf  = (__bf16*)alloc((size_t)Dm * Rr * 2);
    __bf16* Wout_bf = (__bf16*)alloc((size_t)Dm * Dm * 2);
    __bf16* xin_bf  = (__bf16*)alloc((size_t)MTOT * Dm * 2);
    __bf16* zsil_bf = (__bf16*)alloc((size_t)MTOT * Dm * 2);
    float*  xp_f    = (float*) alloc((size_t)MTOT * XPW * 4);
    __bf16* xp_bf   = (__bf16*)alloc((size_t)MTOT * XPW * 2);
    float*  dt_f    = (float*) alloc((size_t)MTOT * Dm * 4);
    __bf16* yg_bf   = (__bf16*)alloc((size_t)MTOT * Dm * 2);

    // 0) f32 -> bf16 staging
    auto conv = [&](const float* s, __bf16* d, int n) {
        k_f32_to_bf16<<<(n + 255) / 256, 256, 0, stream>>>(s, d, n);
    };
    conv(x,     x_bf,    MTOT * Dm);
    conv(W_in,  Win_bf,  2 * Dm * Dm);
    conv(W_x,   Wx_bf,   XPW * Dm);
    conv(W_dt,  Wdt_bf,  Dm * Rr);
    conv(W_out, Wout_bf, Dm * Dm);

    // 1) xz = x @ W_in.T, fused SiLU split
    k_gemm1<<<dim3(2 * Dm / 128, MTOT / 128), 128, 0, stream>>>(x_bf, Win_bf, xin_bf, zsil_bf);
    // 2) xp = xin @ W_x.T
    k_gemm2<<<dim3(1, MTOT / 128), 128, 0, stream>>>(xin_bf, Wx_bf, xp_f, xp_bf);
    // 3) dt = softplus(xp[:, :64] @ W_dt.T + b_dt)
    k_gemm3<<<dim3(Dm / 128, MTOT / 128), 128, 0, stream>>>(xp_bf, Wdt_bf, b_dt, dt_f);
    // 4) selective scan + gating
    k_scan<<<dim3(Dm / 128, BSZ), 128, 0, stream>>>(dt_f, xin_bf, xp_f, zsil_bf, A_log, yg_bf);
    // 5) out = ygated @ W_out.T + b_out
    k_gemm4<<<dim3(Dm / 128, MTOT / 128), 128, 0, stream>>>(yg_bf, Wout_bf, b_out, out);
}
